// DummyConvGRUCell_40346922778953
// MI455X (gfx1250) — compile-verified, matrix operand order checked by reading, not run on hardware
//
#include <hip/hip_runtime.h>

// ---------------------------------------------------------------------------
// EdgeConv-style message passing for MI455X (gfx1250, wave32, WMMA).
//   feat_e = [x[dst], x[src], pos[src]-pos[dst]]  (67 -> padded 96)
//   msg_e  = relu(feat_e @ W + b)                 (f16 WMMA, f32 accum)
//   out[n] = relu(max over edges with dst==n)     (u32 atomicMax, 0-init)
// x (12.8MB) + pos (1.2MB) live in the 192MB L2 -> gathers are L2 hits.
// One lane = one gathered row: 8x global_load_b128 + 4x ds_store_b128 per
// tile instead of scalar fills. Waves fully independent (wave-private LDS
// tiles; DS ops of a wave execute in order), so no barriers in the hot loop.
// ---------------------------------------------------------------------------

typedef __attribute__((ext_vector_type(16))) _Float16 v16h;
typedef __attribute__((ext_vector_type(8)))  _Float16 v8h;
typedef __attribute__((ext_vector_type(4)))  _Float16 v4h;
typedef __attribute__((ext_vector_type(8)))  float    v8f;

#define N_NODES_C 100000
#define N_EDGES_C 1600000
#define CIN_C     32
#define CPOS_C    3
#define COUT_C    32
#define KDIM_C    (2 * CIN_C + CPOS_C)   // 67
#define KPAD_C    96                     // 3 k-blocks of 32
#define WAVES_C   8
#define TPB_C     (WAVES_C * 32)

__global__ void gnn_init_out(float* __restrict__ out, int n) {
    int i = blockIdx.x * blockDim.x + threadIdx.x;
    if (i < n) out[i] = 0.0f;   // u32 pattern 0 == float 0 (atomic-max identity for relu'd msgs)
}

__global__ __launch_bounds__(TPB_C)
void gnn_edge_wmma(const float* __restrict__ x,
                   const float* __restrict__ pos,
                   const int*   __restrict__ ei,      // [2][E]: row0=src(j), row1=dst(i)
                   const float* __restrict__ W,       // [67][32] row-major
                   const float* __restrict__ bias,    // [32]
                   float* __restrict__ out) {         // [N][32], pre-zeroed
    __shared__ alignas(16) _Float16 Wh[KPAD_C][COUT_C];             // 6 KB
    __shared__ alignas(16) _Float16 feat[WAVES_C][16][KPAD_C];      // 24 KB

    // ---- Stage W into LDS as f16, zero-padded rows 67..95 (shared by all waves)
    for (int i = threadIdx.x; i < KPAD_C * COUT_C; i += TPB_C) {
        int k = i / COUT_C, n = i % COUT_C;
        float v = (k < KDIM_C) ? W[k * COUT_C + n] : 0.0f;
        Wh[k][n] = (_Float16)v;
    }
    __syncthreads();   // only cross-wave barrier needed (Wh is read-only after this)

    const int lane = threadIdx.x & 31;
    const int wave = threadIdx.x >> 5;
    const int mh   = lane & 15;   // edge-in-tile / A-C row M / B-C col N
    const int half = lane >> 4;   // K-half selector per ISA f16 A/B layout

    // ---- Build B fragments once; held in VGPRs across the whole edge loop.
    // Per-lane 16-bit fragment layout (CDNA5 ISA):
    //   VGPR v in 0..3 : K = kb*32 + 2v   + half*8  (pairs)
    //   VGPR v in 4..7 : K = kb*32 + 16 + 2(v-4) + half*8
    v16h Bf[3][2];
    #pragma unroll
    for (int kb = 0; kb < 3; ++kb) {
        #pragma unroll
        for (int nt = 0; nt < 2; ++nt) {
            int n = mh + nt * 16;
            v16h bfrag;
            #pragma unroll
            for (int v = 0; v < 4; ++v) {
                int k = kb * 32 + 2 * v + half * 8;
                bfrag[2 * v]     = Wh[k][n];
                bfrag[2 * v + 1] = Wh[k + 1][n];
            }
            #pragma unroll
            for (int v = 4; v < 8; ++v) {
                int k = kb * 32 + 16 + 2 * (v - 4) + half * 8;
                bfrag[2 * v]     = Wh[k][n];
                bfrag[2 * v + 1] = Wh[k + 1][n];
            }
            Bf[kb][nt] = bfrag;
        }
    }
    const float bn0 = bias[mh];        // C/D col n = mh      (nt=0)
    const float bn1 = bias[mh + 16];   // C/D col n = mh + 16 (nt=1)

    // ---- One-time zero padding of columns 64..95 (68..95 stay 0 forever;
    //      64..67 are rewritten with [pos diff, 0] each tile; W rows 67..95
    //      are zero so col 67 content is annihilated regardless).
    #pragma unroll
    for (int e = 0; e < 16; ++e)
        feat[wave][e][64 + lane] = (_Float16)0.0f;

    // ---- Per-wave independent tile loop. E = 16 * 100000 exactly: no ragged tiles.
    const int numTiles  = N_EDGES_C / 16;                 // 100000
    const int waveGlob  = blockIdx.x * WAVES_C + wave;
    const int waveCount = gridDim.x * WAVES_C;

    for (int tile = waveGlob; tile < numTiles; tile += waveCount) {
        const int base = tile * 16;

        // One coalesced load: lanes 0..15 -> src ids, lanes 16..31 -> dst ids.
        const int e0      = base + mh;
        const int idxReg  = (lane < 16) ? ei[e0] : ei[N_EDGES_C + e0];
        // Opposite half's id: lanes 0..15 get dst, lanes 16..31 get src.
        const int rowNode = __shfl_xor(idxReg, 16, 32);

        // ---- Gather: one lane = one 32-float row of x.
        //   lanes 0..15 : x[dst[mh]] -> feat[mh][ 0..31]
        //   lanes 16..31: x[src[mh]] -> feat[mh][32..63]
        // 8x float4 loads walk the 128B row; 4x 16B packed-f16 LDS stores.
        {
            const float*     rowPtr = x + rowNode * CIN_C;
            _Float16* __restrict__ dst16 = &feat[wave][mh][half * 32];
            #pragma unroll
            for (int q = 0; q < 4; ++q) {
                const float4 f0 = *(const float4*)(rowPtr + q * 8);
                const float4 f1 = *(const float4*)(rowPtr + q * 8 + 4);
                v8h h;
                h[0] = (_Float16)f0.x; h[1] = (_Float16)f0.y;
                h[2] = (_Float16)f0.z; h[3] = (_Float16)f0.w;
                h[4] = (_Float16)f1.x; h[5] = (_Float16)f1.y;
                h[6] = (_Float16)f1.z; h[7] = (_Float16)f1.w;
                *(v8h*)(dst16 + q * 8) = h;
            }
        }
        // ---- Pos diff columns 64..66 (+ zeroed 67): lanes 0..15 own one edge each.
        if (lane < 16) {
            const int s = idxReg;    // lanes 0..15 hold src
            const int d = rowNode;   // and fetched dst above
            v4h hp;
            hp[0] = (_Float16)(pos[s * CPOS_C + 0] - pos[d * CPOS_C + 0]);
            hp[1] = (_Float16)(pos[s * CPOS_C + 1] - pos[d * CPOS_C + 1]);
            hp[2] = (_Float16)(pos[s * CPOS_C + 2] - pos[d * CPOS_C + 2]);
            hp[3] = (_Float16)0.0f;
            *(v4h*)&feat[wave][mh][64] = hp;
        }
        // DS ops of one wave execute in order at the LDS; just stop the
        // compiler from reordering the fragment loads above the stores.
        asm volatile("" ::: "memory");

        v8f acc0 = {};
        v8f acc1 = {};
        #pragma unroll
        for (int kb = 0; kb < 3; ++kb) {
            // Two contiguous 16B LDS reads give the whole A fragment
            const v8h lo = *(const v8h*)&feat[wave][mh][kb * 32 + half * 8];
            const v8h hi = *(const v8h*)&feat[wave][mh][kb * 32 + 16 + half * 8];
            v16h a = __builtin_shufflevector(lo, hi,
                0, 1, 2, 3, 4, 5, 6, 7, 8, 9, 10, 11, 12, 13, 14, 15);
            acc0 = __builtin_amdgcn_wmma_f32_16x16x32_f16(
                false, a, false, Bf[kb][0], (short)0, acc0, false, false);
            acc1 = __builtin_amdgcn_wmma_f32_16x16x32_f16(
                false, a, false, Bf[kb][1], (short)0, acc1, false, false);
        }
        asm volatile("" ::: "memory");   // keep next tile's stores below these loads

        // ---- Epilogue: bias + relu + scatter-max (branch-free, always full tile).
        // C/D layout: vgpr r, lane = n + 16*(m>>3)  ->  m = half*8 + r, n = mh (+16).
        // relu'd values are >= 0, so float max == u32-bitpattern max.
        #pragma unroll
        for (int r = 0; r < 8; ++r) {
            const int m    = half * 8 + r;
            const int node = __shfl(idxReg, 16 + m, 32);   // dst id for row m
            const float v0 = fmaxf(acc0[r] + bn0, 0.0f);
            const float v1 = fmaxf(acc1[r] + bn1, 0.0f);
            atomicMax((unsigned int*)&out[node * COUT_C + mh],
                      __float_as_uint(v0));
            atomicMax((unsigned int*)&out[node * COUT_C + 16 + mh],
                      __float_as_uint(v1));
        }
    }
}

extern "C" void kernel_launch(void* const* d_in, const int* in_sizes, int n_in,
                              void* d_out, int out_size, void* d_ws, size_t ws_size,
                              hipStream_t stream) {
    // setup_inputs order: 0:h 1:x 2:pos 3:edge_index_gate 4:edge_index_cand 5:W 6:b
    const float* x    = (const float*)d_in[1];
    const float* pos  = (const float*)d_in[2];
    const int*   ei   = (const int*)d_in[4];   // edge_index_cand (gate is unused)
    const float* W    = (const float*)d_in[5];
    const float* bias = (const float*)d_in[6];
    float*       out  = (float*)d_out;

    (void)in_sizes; (void)n_in; (void)d_ws; (void)ws_size;

    gnn_init_out<<<(out_size + 255) / 256, 256, 0, stream>>>(out, out_size);

    const int numTiles = N_EDGES_C / 16;                         // 100000
    const int numWGs   = (numTiles + WAVES_C - 1) / WAVES_C;     // 12500 -> 1 tile/wave
    gnn_edge_wmma<<<numWGs, TPB_C, 0, stream>>>(x, pos, ei, W, bias, out);
}
